// MyRNN_69071664054578
// MI455X (gfx1250) — matrix operand order repaired and checked by reading, not compiled, not run on hardware
//
#include <hip/hip_runtime.h>

typedef __attribute__((ext_vector_type(16))) _Float16 v16h;
typedef __attribute__((ext_vector_type(4)))  _Float16 v4h;
typedef __attribute__((ext_vector_type(8)))  float    v8f;
typedef __attribute__((ext_vector_type(4)))  float    v4f;

#define SEQ   2048
#define FEAT  64
#define UNITS 64
#define OUTF  6

static __device__ __forceinline__ float sigm(float x) { return 1.f / (1.f + __expf(-x)); }
static __device__ __forceinline__ float tanhx(float x) {
  float e = __expf(-2.f * x);
  return (1.f - e) / (1.f + e);
}

// ---------------------------------------------------------------------------
// Prepack [W;U] (128x256 f32) -> f16 WMMA B-fragments.
// Fragment index layout: Bp[((cell*16 + ntile)*4 + kchunk)*32 + lane][16 halves]
// B (32x16, f16) lane layout chosen here, same k order as A: lanes 0-15 hold K=0..15 (halves in order),
// lanes 16-31 hold K=16..31, column N = lane % 16.
// ---------------------------------------------------------------------------
__global__ void lstm_prepack(const float* __restrict__ W0, const float* __restrict__ U0,
                             const float* __restrict__ W1, const float* __restrict__ U1,
                             _Float16* __restrict__ Bp) {
  int idx  = blockIdx.x * blockDim.x + threadIdx.x;   // 65536 total
  int hh   = idx & 15;
  int lane = (idx >> 4) & 31;
  int kc   = (idx >> 9) & 3;
  int nt   = (idx >> 11) & 15;
  int cw   = (idx >> 15) & 1;
  int kfrag = hh + ((lane >= 16) ? 16 : 0);
  int krow  = kc * 32 + kfrag;                        // row in [W;U] (128 rows)
  int col   = nt * 16 + (lane & 15);                  // col in 4*UNITS (256)
  const float* Wm = cw ? W1 : W0;
  const float* Um = cw ? U1 : U0;
  float v = (krow < 64) ? Wm[krow * 256 + col] : Um[(krow - 64) * 256 + col];
  Bp[idx] = (_Float16)v;
}

// ---------------------------------------------------------------------------
// Persistent recurrence kernel: one workgroup = 16 batch rows, 8 waves.
// Per step, per cell: z[16,256] = [in|h] (16x128, f16 in LDS A-frag layout)
//                                 @ [W;U] (128x256, f16 B-frags in VGPRs)
// 8 WMMAs per wave per cell; x_{t+1} load software-pipelined across the step.
// ---------------------------------------------------------------------------
__global__ __launch_bounds__(256, 1)
void lstm_persist(const float* __restrict__ x,
                  const float* __restrict__ b0g, const float* __restrict__ b1g,
                  const float* __restrict__ Wf, const float* __restrict__ bf,
                  const float* __restrict__ Wo, const float* __restrict__ bo,
                  const _Float16* __restrict__ Bp,
                  float* __restrict__ out) {
  __shared__ __align__(32) _Float16 Abuf[3][4][32][16]; // [cell][kchunk][lane][half]
  __shared__ float zbuf[16][256];
  __shared__ float hplain[16][64];

  const int tid   = threadIdx.x;
  const int lane  = tid & 31;
  const int wave  = tid >> 5;
  const int m     = tid & 15;          // batch row within tile (elementwise phases)
  const int jb    = (tid >> 4) << 2;   // 4 gate-columns j = jb..jb+3
  const int kk    = jb & 31;           // K within a 32-wide A chunk
  const int kcl   = jb >> 5;           // which 32-chunk of the 64-wide half
  const int hh0   = kk & 15;                         // A-frag half index
  const int laneW = m + ((kk & 16) ? 16 : 0);        // A-frag lane

  // ---- loop-invariant weight B-fragments into registers ----
  v16h bw[2][2][4];   // [weight-set][ntile(wave,wave+8)][kchunk]
#pragma unroll
  for (int s = 0; s < 2; ++s)
#pragma unroll
    for (int ti = 0; ti < 2; ++ti)
#pragma unroll
      for (int kc = 0; kc < 4; ++kc) {
        int nt = wave + ti * 8;
        size_t off = ((((size_t)s * 16 + nt) * 4 + kc) * 32 + lane) * 16;
        bw[s][ti][kc] = *(const v16h*)(Bp + off);
      }

  // ---- per-thread bias registers (gate order i,f,g,o) ----
  v4f bia[2][4];
#pragma unroll
  for (int g = 0; g < 4; ++g) {
    bia[0][g] = *(const v4f*)(b0g + g * 64 + jb);
    bia[1][g] = *(const v4f*)(b1g + g * 64 + jb);
  }

  // ---- zero initial hidden state staging ----
  for (int i = tid; i < 3 * 4 * 32 * 16; i += 256)
    ((_Float16*)Abuf)[i] = (_Float16)0.f;
  __syncthreads();

  float cst[3][4] = {};   // cell state c, register-resident across all steps

  const float* xrow = x + (size_t)(blockIdx.x * 16 + m) * (SEQ * FEAT) + jb;

  // ---- prologue: stage x_0 into cell-0 A-fragment slots (K=0..63) ----
  {
    v4f xv = *(const v4f*)(xrow);
    v4h xh;
#pragma unroll
    for (int r = 0; r < 4; ++r) xh[r] = (_Float16)xv[r];
    *(v4h*)(&Abuf[0][kcl][laneW][hh0]) = xh;
  }
  __syncthreads();

  for (int t = 0; t < SEQ; ++t) {
    // Issue next step's x load now; its latency hides behind 3 cells of work.
    const int tn = (t + 1 < SEQ) ? t + 1 : t;
    v4f xnext = *(const v4f*)(xrow + (size_t)tn * FEAT);
    // Deep prefetch into caches (no LOADcnt cost).
    const int tp = (t + 8 < SEQ) ? t + 8 : t;
    __builtin_prefetch(xrow + (size_t)tp * FEAT, 0, 3);

#pragma unroll
    for (int cell = 0; cell < 3; ++cell) {
      const int wsel = (cell == 0) ? 0 : 1;   // layers 2&3 share weight set 1

      // A fragments: [input(64) | h_prev(64)] = 16x128
      v16h a[4];
#pragma unroll
      for (int kc = 0; kc < 4; ++kc)
        a[kc] = *(const v16h*)(&Abuf[cell][kc][lane][0]);

      v8f acc0 = {};
      v8f acc1 = {};
#pragma unroll
      for (int kc = 0; kc < 4; ++kc) {
        acc0 = __builtin_amdgcn_wmma_f32_16x16x32_f16(
            false, a[kc], false, bw[wsel][0][kc], (short)0, acc0, false, false);
        acc1 = __builtin_amdgcn_wmma_f32_16x16x32_f16(
            false, a[kc], false, bw[wsel][1][kc], (short)0, acc1, false, false);
      }

      // scatter z (C layout: lanes 0-15 -> M=v, lanes 16-31 -> M=v+8; N=lane%16)
      {
        const int mr = (lane < 16) ? 0 : 8;
        const int c0 = wave * 16 + (lane & 15);
        const int c1 = (wave + 8) * 16 + (lane & 15);
#pragma unroll
        for (int v = 0; v < 8; ++v) {
          zbuf[mr + v][c0] = acc0[v];
          zbuf[mr + v][c1] = acc1[v];
        }
      }
      __syncthreads();

      // ---- elementwise LSTM gates; c stays in registers ----
      v4f zi = *(const v4f*)(&zbuf[m][jb]);
      v4f zf = *(const v4f*)(&zbuf[m][64 + jb]);
      v4f zg = *(const v4f*)(&zbuf[m][128 + jb]);
      v4f zo = *(const v4f*)(&zbuf[m][192 + jb]);
      v4h hv;
#pragma unroll
      for (int r = 0; r < 4; ++r) {
        float iv = sigm(zi[r] + bia[wsel][0][r]);
        float fv = sigm(zf[r] + bia[wsel][1][r]);
        float gv = tanhx(zg[r] + bia[wsel][2][r]);
        float ov = sigm(zo[r] + bia[wsel][3][r]);
        float c  = fv * cst[cell][r] + iv * gv;
        cst[cell][r] = c;
        hv[r] = (_Float16)(ov * tanhx(c));
      }
      // own h_prev slot (K=64..127) for step t+1
      *(v4h*)(&Abuf[cell][2 + kcl][laneW][hh0]) = hv;
      if (cell < 2) {
        // input slot (K=0..63) of the next cell this step
        *(v4h*)(&Abuf[cell + 1][kcl][laneW][hh0]) = hv;
      } else {
        // stage x_{t+1} into cell-0 input slots (safe: cell-0 A reads for step
        // t completed several barriers ago; next read is after the barrier below)
        v4h xh;
#pragma unroll
        for (int r = 0; r < 4; ++r) xh[r] = (_Float16)xnext[r];
        *(v4h*)(&Abuf[0][kcl][laneW][hh0]) = xh;
        if (t == SEQ - 1) {  // h2 needed in plain layout only at the end
#pragma unroll
          for (int r = 0; r < 4; ++r) hplain[m][jb + r] = (float)hv[r];
        }
      }
      __syncthreads();
    }
  }

  // ---- dense heads: y = relu(h2 @ Wf + bf) @ Wo + bo (tiny, VALU) ----
  v4f fa = *(const v4f*)(bf + jb);
#pragma unroll 8
  for (int k = 0; k < UNITS; ++k) {
    float hk = hplain[m][k];
    v4f wr = *(const v4f*)(Wf + k * 64 + jb);
#pragma unroll
    for (int r = 0; r < 4; ++r) fa[r] += hk * wr[r];
  }
#pragma unroll
  for (int r = 0; r < 4; ++r) zbuf[m][jb + r] = fmaxf(fa[r], 0.f);
  __syncthreads();

  if (tid < 16 * OUTF) {
    int m2 = tid / OUTF, o = tid % OUTF;
    float s = bo[o];
    for (int k = 0; k < UNITS; ++k) s += zbuf[m2][k] * Wo[k * OUTF + o];
    out[(blockIdx.x * 16 + m2) * OUTF + o] = s;
  }
}

extern "C" void kernel_launch(void* const* d_in, const int* in_sizes, int n_in,
                              void* d_out, int out_size, void* d_ws, size_t ws_size,
                              hipStream_t stream) {
  (void)in_sizes; (void)n_in; (void)out_size; (void)ws_size;
  const float* x  = (const float*)d_in[0];
  const float* W0 = (const float*)d_in[1];
  const float* U0 = (const float*)d_in[2];
  const float* b0 = (const float*)d_in[3];
  const float* W1 = (const float*)d_in[4];
  const float* U1 = (const float*)d_in[5];
  const float* b1 = (const float*)d_in[6];
  const float* Wf = (const float*)d_in[7];
  const float* bf = (const float*)d_in[8];
  const float* Wo = (const float*)d_in[9];
  const float* bo = (const float*)d_in[10];

  _Float16* Bp = (_Float16*)d_ws;   // 131072 bytes of prepacked weight fragments

  lstm_prepack<<<256, 256, 0, stream>>>(W0, U0, W1, U1, Bp);
  lstm_persist<<<8, 256, 0, stream>>>(x, b0, b1, Wf, bf, Wo, bo, Bp, (float*)d_out);
}
